// GCN_61014305407058
// MI455X (gfx1250) — compile-verified, matrix operand order checked by reading, not run on hardware
//
#include <hip/hip_runtime.h>

// ---------------- problem sizes ----------------
#define N_NODES (512 * 784)          // 401408
#define N_EDGES (8 * N_NODES)        // 3211264
#define NPG 784
#define K_FC1 (NPG * 64)             // 50176
#define B_GRAPHS 512

typedef _Float16 h16;
typedef __attribute__((ext_vector_type(16))) _Float16 v16h;
typedef __attribute__((ext_vector_type(2)))  _Float16 v2h;
typedef __attribute__((ext_vector_type(8)))  float    v8f;

// ---------------- helper kernels ----------------
__global__ void k_zero_i32(int* __restrict__ p, int n) {
    int i = blockIdx.x * blockDim.x + threadIdx.x;
    if (i < n) p[i] = 0;
}

// degree count (in-degree); self loop added later as +1
__global__ void k_count(const long long* __restrict__ ei, int* __restrict__ cnt) {
    int e = blockIdx.x * blockDim.x + threadIdx.x;
    if (e < N_EDGES) {
        int d = (int)ei[N_EDGES + e];
        atomicAdd(&cnt[d], 1);
    }
}

__global__ void k_dinv(const int* __restrict__ cnt, float* __restrict__ dinv) {
    int n = blockIdx.x * blockDim.x + threadIdx.x;
    if (n < N_NODES) dinv[n] = rsqrtf(1.0f + (float)cnt[n]);
}

// layer-1 scalar aggregation: a1[n] = x[n]*dinv[n]^2 (self loop), then edge scatter
__global__ void k_a1_init(const float* __restrict__ x, const float* __restrict__ dinv,
                          float* __restrict__ a1) {
    int n = blockIdx.x * blockDim.x + threadIdx.x;
    if (n < N_NODES) a1[n] = x[n] * dinv[n] * dinv[n];
}

__global__ void k_a1_edges(const long long* __restrict__ ei, const float* __restrict__ x,
                           const float* __restrict__ dinv, float* __restrict__ a1) {
    int e = blockIdx.x * blockDim.x + threadIdx.x;
    if (e < N_EDGES) {
        int s = (int)ei[e];
        int d = (int)ei[N_EDGES + e];
        atomicAdd(&a1[d], x[s] * dinv[s] * dinv[d]);
    }
}

// out1[n,f] = relu(a1[n]*W1[f] + b1[f]),  f in [0,32)
__global__ void k_out1(const float* __restrict__ a1, const float* __restrict__ W1,
                       const float* __restrict__ b1, float* __restrict__ out1) {
    int i = blockIdx.x * blockDim.x + threadIdx.x;
    if (i < N_NODES * 32) {
        int f = i & 31;
        int n = i >> 5;
        float v = a1[n] * W1[f] + b1[f];
        out1[i] = v > 0.0f ? v : 0.0f;
    }
}

// layer-2 aggregation over the 32-dim out1 (W2 applied AFTER aggregation: A(HW)=(AH)W)
__global__ void k_agg2_init(const float* __restrict__ out1, const float* __restrict__ dinv,
                            float* __restrict__ agg2) {
    int i = blockIdx.x * blockDim.x + threadIdx.x;
    if (i < N_NODES * 32) {
        int n = i >> 5;
        agg2[i] = out1[i] * dinv[n] * dinv[n];
    }
}

// one wave32 per edge, one lane per feature
__global__ void k_agg2_edges(const long long* __restrict__ ei, const float* __restrict__ out1,
                             const float* __restrict__ dinv, float* __restrict__ agg2) {
    int lane = threadIdx.x & 31;
    int e = blockIdx.x * (blockDim.x >> 5) + (threadIdx.x >> 5);
    if (e < N_EDGES) {
        int s = (int)ei[e];
        int d = (int)ei[N_EDGES + e];
        float coef = dinv[s] * dinv[d];
        atomicAdd(&agg2[d * 32 + lane], out1[s * 32 + lane] * coef);
    }
}

// feat[n,j] = relu(sum_i agg2[n,i]*W2[i,j] + b2[j]) stored as f16 (FC1 A operand)
// block = 4 nodes x 64 feats; agg2 rows + W2 staged through LDS
__global__ void __launch_bounds__(256) k_feat(const float* __restrict__ agg2,
                                              const float* __restrict__ W2,
                                              const float* __restrict__ b2,
                                              h16* __restrict__ feat) {
    __shared__ float s_w2[32 * 64];
    __shared__ float s_a[4 * 32];
    int tid = threadIdx.x;
    for (int i = tid; i < 32 * 64; i += 256) s_w2[i] = W2[i];
    if (tid < 128) {
        int r = tid >> 5, c = tid & 31;
        s_a[tid] = agg2[((long long)blockIdx.x * 4 + r) * 32 + c];
    }
    __syncthreads();
    int ln = tid >> 6;
    int j  = tid & 63;
    long long node = (long long)blockIdx.x * 4 + ln;
    float s = b2[j];
#pragma unroll
    for (int i = 0; i < 32; ++i) s += s_a[ln * 32 + i] * s_w2[i * 64 + j];
    feat[node * 64 + j] = (h16)(s > 0.0f ? s : 0.0f);
}

// transpose + convert fc1_w [K,128] f32 -> wt [128][K] f16 (contiguous K per out column)
__global__ void k_wt(const float* __restrict__ w, h16* __restrict__ wt) {
    long long i = (long long)blockIdx.x * blockDim.x + threadIdx.x;
    if (i < (long long)K_FC1 * 128) {
        int n = (int)(i & 127);
        long long k = i >> 7;
        wt[(long long)n * K_FC1 + k] = (h16)w[i];
    }
}

// FC1 GEMM: Y[512,128] = relu(X[512,50176] @ W[50176,128] + b) via v_wmma_f32_16x16x32_f16
// grid = 32 M-tiles; block = 256 = 8 waves; wave w owns output columns [16w, 16w+16)
__global__ void __launch_bounds__(256) k_fc1(const h16* __restrict__ X,
                                             const h16* __restrict__ Wt,
                                             const float* __restrict__ bias,
                                             float* __restrict__ Y) {
    int wave = threadIdx.x >> 5;
    int lane = threadIdx.x & 31;
    int m16  = lane & 15;
    int hsel = lane >> 4;           // 0: K 0-7 and 16-23, 1: K 8-15 and 24-31 (per ISA A/B layout)
    int n0   = wave * 16;

    const h16* arow = X  + (long long)(blockIdx.x * 16 + m16) * K_FC1; // A row = graph m
    const h16* brow = Wt + (long long)(n0 + m16) * K_FC1;              // B col = out col n

    v8f acc = {};
    for (int k0 = 0; k0 < K_FC1; k0 += 32) {
        v16h a, b;
#pragma unroll
        for (int q = 0; q < 8; ++q) {
            // ISA 16-bit A/B fragment: VGPR q holds K = (q%4)*2 + (q/4)*16 + hsel*8, +1
            int k = k0 + ((q & 3) << 1) + ((q >> 2) << 4) + (hsel << 3);
            v2h pa = *(const v2h*)(arow + k);
            v2h pb = *(const v2h*)(brow + k);
            a[2 * q]     = pa[0];
            a[2 * q + 1] = pa[1];
            b[2 * q]     = pb[0];
            b[2 * q + 1] = pb[1];
        }
        if (k0 + 288 <= K_FC1) {
            __builtin_prefetch(arow + k0 + 256, 0, 1);  // global_prefetch_b8
            __builtin_prefetch(brow + k0 + 256, 0, 1);
        }
        acc = __builtin_amdgcn_wmma_f32_16x16x32_f16(
            /*neg_a=*/false, a, /*neg_b=*/false, b,
            /*c_mod=*/(short)0, acc, /*reuse_a=*/false, /*reuse_b=*/false);
    }

    // f32 C/D layout: lanes 0-15 -> M=r, lanes 16-31 -> M=8+r; N = lane%16
    int ncol = n0 + (lane & 15);
    float bv = bias[ncol];
#pragma unroll
    for (int r = 0; r < 8; ++r) {
        int mrow = blockIdx.x * 16 + hsel * 8 + r;
        float v = acc[r] + bv;
        Y[mrow * 128 + ncol] = v > 0.0f ? v : 0.0f;
    }
}

// FC2: out[512,10]
__global__ void k_fc2(const float* __restrict__ fc1, const float* __restrict__ w,
                      const float* __restrict__ b, float* __restrict__ out) {
    int i = blockIdx.x * blockDim.x + threadIdx.x;
    if (i < B_GRAPHS * 10) {
        int g = i / 10, c = i % 10;
        float s = b[c];
#pragma unroll 8
        for (int j = 0; j < 128; ++j) s += fc1[g * 128 + j] * w[j * 10 + c];
        out[i] = s;
    }
}

// ---------------- launcher ----------------
extern "C" void kernel_launch(void* const* d_in, const int* in_sizes, int n_in,
                              void* d_out, int out_size, void* d_ws, size_t ws_size,
                              hipStream_t stream) {
    const float*     x    = (const float*)d_in[0];
    const long long* ei   = (const long long*)d_in[1];   // int64 [2, E]
    const float*     W1   = (const float*)d_in[2];
    const float*     b1   = (const float*)d_in[3];
    const float*     W2   = (const float*)d_in[4];
    const float*     b2   = (const float*)d_in[5];
    const float*     fc1w = (const float*)d_in[6];
    const float*     fc1b = (const float*)d_in[7];
    const float*     fc2w = (const float*)d_in[8];
    const float*     fc2b = (const float*)d_in[9];
    float* out = (float*)d_out;

    char* ws = (char*)d_ws;
    size_t off = 0;
    float* dinv = (float*)(ws + off);             off += (size_t)N_NODES * 4;
    float* a1   = (float*)(ws + off);
    int*   cnt  = (int*)a1;                       off += (size_t)N_NODES * 4;
    float* out1 = (float*)(ws + off);             // reused as feat_h (same byte size)
    h16*   feat = (h16*)out1;                     off += (size_t)N_NODES * 32 * 4;
    float* agg2 = (float*)(ws + off);             off += (size_t)N_NODES * 32 * 4;
    h16*   wt   = (h16*)(ws + off);               off += (size_t)K_FC1 * 128 * 2;
    float* fc1o = (float*)(ws + off);             off += (size_t)B_GRAPHS * 128 * 4;

    const int T = 256;

    // degree / D^{-1/2}
    k_zero_i32<<<(N_NODES + T - 1) / T, T, 0, stream>>>(cnt, N_NODES);
    k_count<<<(N_EDGES + T - 1) / T, T, 0, stream>>>(ei, cnt);
    k_dinv<<<(N_NODES + T - 1) / T, T, 0, stream>>>(cnt, dinv);

    // fc1_w transpose+convert (independent, kick it off early)
    k_wt<<<(int)(((long long)K_FC1 * 128 + T - 1) / T), T, 0, stream>>>(fc1w, wt);

    // GCN layer 1 (scalar aggregation, then expand by W1)
    k_a1_init<<<(N_NODES + T - 1) / T, T, 0, stream>>>(x, dinv, a1);
    k_a1_edges<<<(N_EDGES + T - 1) / T, T, 0, stream>>>(ei, x, dinv, a1);
    k_out1<<<(N_NODES * 32 + T - 1) / T, T, 0, stream>>>(a1, W1, b1, out1);

    // GCN layer 2 (aggregate 32-dim, then apply W2)
    k_agg2_init<<<(N_NODES * 32 + T - 1) / T, T, 0, stream>>>(out1, dinv, agg2);
    k_agg2_edges<<<(N_EDGES / 8), T, 0, stream>>>(ei, out1, dinv, agg2);
    k_feat<<<(N_NODES / 4), T, 0, stream>>>(agg2, W2, b2, feat);

    // FC1 via WMMA, then FC2
    k_fc1<<<B_GRAPHS / 16, T, 0, stream>>>(feat, wt, fc1b, fc1o);
    k_fc2<<<(B_GRAPHS * 10 + T - 1) / T, T, 0, stream>>>(fc1o, fc2w, fc2b, out);
}
